// ScRRAMBLeCapsLayer_82789789597931
// MI455X (gfx1250) — matrix operand order repaired and checked by reading, not compiled, not run on hardware
//
#include <hip/hip_runtime.h>
#include <hip/hip_bf16.h>

// Problem constants (from reference)
#define IN_SIZE      262144
#define NCAPS        1024
#define CAPSZ        256
#define RF           64
#define RFPC         4                  // RF_PER_CAP
#define SRC_D        4096               // INPUT_EFF_CAPS * RFPC (flattened source index)
#define DST_D        4096               // NUM_CAPSULES * RFPC  (flattened dest index)

typedef __attribute__((ext_vector_type(16))) _Float16 v16h;
typedef __attribute__((ext_vector_type(8)))  _Float16 v8h;
typedef __attribute__((ext_vector_type(2)))  _Float16 h2;
typedef __attribute__((ext_vector_type(8)))  float    v8f;
typedef __attribute__((ext_vector_type(4)))  float    f4;

// -----------------------------------------------------------------------------
// Stage 1: x_routed[d, m] = sum_s Ci[s, d] * x[s, m]
//   Ci flat:  Ci[s*4096 + d]   (s = i*4+j, d = k*4+l)
//   x  flat:  x[s*64 + m]
//   out flat: xr[d*64 + m]  (== xr[cap*256 + k*64 + m])
//
// One block per 16-wide d-tile; 4 waves, wave w owns m-tile [16w,16w+16).
// LDS tiles are stored TRANSPOSED ([row][kk]) so each lane's 16 f16 operand
// elements are two contiguous 16B runs -> 2x ds_load_b128 per operand,
// no scalar u16 loads / v_mov_b16 packing in the WMMA loop.
// Row stride = 40 halves (80B): 16B-aligned rows, conflict-free row starts.
// -----------------------------------------------------------------------------
__global__ __launch_bounds__(128)
void scrramble_route_wmma(const float* __restrict__ x,
                          const float* __restrict__ Ci,
                          float* __restrict__ xr) {
    __shared__ _Float16 AsT[16][40];   // [dd][kk]
    __shared__ _Float16 BsT[64][40];   // [m][kk]

    const int d0   = blockIdx.x * 16;
    const int tid  = threadIdx.x;          // 0..127
    const int lane = tid & 31;
    const int wave = tid >> 5;             // 0..3
    const int m0   = wave * 16;

    // Lane decode per CDNA5 WMMA VGPR layouts (wave32):
    // A (16x32 f16): lanes 0-15: M=lane,    elems 0..7 = K 0..7,  8..15 = K 16..23
    //                lanes16-31: M=lane-16, elems 0..7 = K 8..15, 8..15 = K 24..31
    const int a_col = lane & 15;           // dd (row M)
    const int a_kb  = (lane >> 4) * 8;     // K base (0 or 8)
    // B (32x16 f16): lanes 0-15: N=lane,    elems 0..7 = K 0..7,  8..15 = K 16..23
    //                lanes16-31: N=lane-16, elems 0..7 = K 8..15, 8..15 = K 24..31
    const int b_row = m0 + (lane & 15);    // m (col N)
    const int b_kb  = (lane >> 4) * 8;     // K base (0 or 8)

    v8f acc = {};

    for (int k0 = 0; k0 < SRC_D; k0 += 32) {
        // --- Ci block (32 k-rows x 16 d-cols): threads 0..63, k-row pairs,
        //     packed (k,k+1) f16 pairs stored as aligned dwords into AsT.
        if (tid < 64) {
            const int kx = (tid >> 2) * 2;         // 0,2,..,30
            const int cc = (tid & 3) * 4;          // 0,4,8,12
            const f4 r0 = *(const f4*)(Ci + (size_t)(k0 + kx    ) * SRC_D + d0 + cc);
            const f4 r1 = *(const f4*)(Ci + (size_t)(k0 + kx + 1) * SRC_D + d0 + cc);
            #pragma unroll
            for (int i = 0; i < 4; ++i) {
                h2 p = { (_Float16)r0[i], (_Float16)r1[i] };
                *(h2*)&AsT[cc + i][kx] = p;
            }
        }
        // --- x block (32 k-rows x 64 m-cols): all 128 threads, 2 row-pair
        //     float4 loads each, packed dword stores into BsT.
        #pragma unroll
        for (int r = 0; r < 2; ++r) {
            const int idx = r * 128 + tid;         // 0..255
            const int kx  = (idx >> 4) * 2;        // 0,2,..,30
            const int mx  = (idx & 15) * 4;        // 0,4,..,60
            const f4 r0 = *(const f4*)(x + (size_t)(k0 + kx    ) * RF + mx);
            const f4 r1 = *(const f4*)(x + (size_t)(k0 + kx + 1) * RF + mx);
            #pragma unroll
            for (int i = 0; i < 4; ++i) {
                h2 p = { (_Float16)r0[i], (_Float16)r1[i] };
                *(h2*)&BsT[mx + i][kx] = p;
            }
        }
        __syncthreads();

        // --- operand fetch: 2x ds_load_b128 per operand, then WMMA ---
        const v8h a0 = *(const v8h*)&AsT[a_col][a_kb];          // K a_kb..a_kb+7
        const v8h a1 = *(const v8h*)&AsT[a_col][16 + a_kb];     // K 16+a_kb..
        const v8h b0 = *(const v8h*)&BsT[b_row][b_kb];          // K b_kb..b_kb+7
        const v8h b1 = *(const v8h*)&BsT[b_row][16 + b_kb];     // K 16+b_kb..
        const v16h a = __builtin_shufflevector(a0, a1, 0,1,2,3,4,5,6,7,8,9,10,11,12,13,14,15);
        const v16h b = __builtin_shufflevector(b0, b1, 0,1,2,3,4,5,6,7,8,9,10,11,12,13,14,15);

        acc = __builtin_amdgcn_wmma_f32_16x16x32_f16(
            /*neg_a=*/false, a, /*neg_b=*/false, b,
            /*c_mod=*/(short)0, acc, /*reuse_a=*/false, /*reuse_b=*/false);

        __syncthreads();
    }

    // --- store 16x16 f32 tile; C/D layout: VGPR g -> row g + 8*(lane>=16), col lane&15
    const int row_b = (lane >> 4) * 8;
    const int mm    = m0 + (lane & 15);
    #pragma unroll
    for (int g = 0; g < 8; ++g) {
        xr[(size_t)(d0 + row_b + g) * RF + mm] = acc[g];
    }
}

// -----------------------------------------------------------------------------
// Stage 2: y[cap, j, l] = sum_{k,m} Wi[cap,j,k,l,m] * xr[cap*256 + k*64 + m]
// One thread per output element. Wi (268 MB, streamed once) uses non-temporal
// loads to avoid evicting the L2-resident xr/x/Ci working set; each wave fully
// consumes a contiguous 8KB Wi region over its m-loop.
// -----------------------------------------------------------------------------
__global__ __launch_bounds__(256)
void scrramble_caps_gemv(const float* __restrict__ Wi,
                         const float* __restrict__ xr,
                         float* __restrict__ y) {
    const int o   = blockIdx.x * blockDim.x + threadIdx.x;   // 0..262143
    const int cap = o >> 8;
    const int j   = (o >> 6) & 3;
    const int l   = o & 63;

    const float* __restrict__ w = Wi + (size_t)cap * 65536 + (size_t)j * 16384 + (size_t)l * 64;
    const float* __restrict__ v = xr + (size_t)cap * 256;

    float s = 0.0f;
    #pragma unroll
    for (int k = 0; k < RFPC; ++k) {
        const f4* wr = (const f4*)(w + (size_t)k * 4096);
        const f4* vr = (const f4*)(v + k * 64);
        #pragma unroll
        for (int q = 0; q < 16; ++q) {
            const f4 a  = __builtin_nontemporal_load(wr + q);
            const f4 bb = vr[q];
            s = fmaf(a.x, bb.x, s);
            s = fmaf(a.y, bb.y, s);
            s = fmaf(a.z, bb.z, s);
            s = fmaf(a.w, bb.w, s);
        }
    }
    y[o] = s;
}

extern "C" void kernel_launch(void* const* d_in, const int* in_sizes, int n_in,
                              void* d_out, int out_size, void* d_ws, size_t ws_size,
                              hipStream_t stream) {
    const float* x  = (const float*)d_in[0];   // (262144,)            f32
    const float* Ci = (const float*)d_in[1];   // (1024,4,1024,4)      f32
    const float* Wi = (const float*)d_in[2];   // (1024,4,4,64,64)     f32
    float* y  = (float*)d_out;                 // (1024,4,64)          f32
    float* xr = (float*)d_ws;                  // scratch: 4096*64 f32 = 1 MB

    // Stage 1: routing GEMM (4096x4096x64) via f16 WMMA, f32 accumulate.
    scrramble_route_wmma<<<DST_D / 16, 128, 0, stream>>>(x, Ci, xr);

    // Stage 2: 1024 independent 256x256 mat-vecs (Wi streaming bound).
    scrramble_caps_gemv<<<(NCAPS * CAPSZ) / 256, 256, 0, stream>>>(Wi, xr, y);
}
